// DeltaNet_31877247271463
// MI455X (gfx1250) — compile-verified, matrix operand order checked by reading, not run on hardware
//
#include <hip/hip_runtime.h>
#include <math.h>

typedef __bf16 bf16;
typedef __attribute__((ext_vector_type(16))) __bf16 v16bf;
typedef __attribute__((ext_vector_type(8)))  float  v8f;

#define NHEAD 4
#define DHEAD 256
#define HIDSZ 1024
#define GATEK 1040
#define FLOORW 0.05f

// ---------------------------------------------------------------------------
// WMMA 16x16x32 bf16 (wave32). Operands are staged in LDS in *fragment order*:
// per 16x32 subtile, lane l owns 16 contiguous bf16 (32 bytes):
//   A: lane = (m&15) + 16*((kw>>3)&1), slot = (kw&7) + 8*((kw>>4)&1)
//   B: lane = (n&15) + 16*((kw>>3)&1), slot = (kw&7) + 8*((kw>>4)&1)
// so a fragment load is a single aligned 32-byte LDS read (2x ds_load_b128).
// ---------------------------------------------------------------------------
__device__ __forceinline__ v8f wmma_bf16(v16bf a, v16bf b, v8f c) {
  return __builtin_amdgcn_wmma_f32_16x16x32_bf16(false, a, false, b, (short)0, c,
                                                 false, false);
}

__device__ __forceinline__ int swz_addrA(int row, int k, int KT) {
  int kt = k >> 5, kw = k & 31;
  return (((row >> 4) * KT + kt) * 32 + (row & 15) + (((kw >> 3) & 1) << 4)) * 16 +
         (kw & 7) + ((kw >> 4) << 3);
}
__device__ __forceinline__ int swz_addrB(int k, int n, int NT) {
  int kt = k >> 5, kw = k & 31;
  return ((kt * NT + (n >> 4)) * 32 + (n & 15) + (((kw >> 3) & 1) << 4)) * 16 +
         (kw & 7) + ((kw >> 4) << 3);
}
__device__ __forceinline__ v16bf ld_frag(const bf16* p) {
  return *(const v16bf*)p;
}

// ---------------------------------------------------------------------------
// fp32 -> bf16 conversion (plain, and transposing for GEMM B operands)
// ---------------------------------------------------------------------------
__global__ void f2bf_kernel(const float* __restrict__ in, bf16* __restrict__ out,
                            size_t n) {
  size_t i = (size_t)blockIdx.x * blockDim.x + threadIdx.x;
  size_t stride = (size_t)gridDim.x * blockDim.x;
  for (; i < n; i += stride) out[i] = (bf16)in[i];
}

__global__ void f2bf_t_kernel(const float* __restrict__ in, bf16* __restrict__ out,
                              int K, int N) {  // in: K x N  ->  out: N x K
  size_t idx = (size_t)blockIdx.x * blockDim.x + threadIdx.x;
  if (idx >= (size_t)K * N) return;
  int k = (int)(idx / N), n = (int)(idx % N);
  out[(size_t)n * K + k] = (bf16)in[idx];
}

// ---------------------------------------------------------------------------
// bf16 WMMA GEMM: C(MxN,f32) = A(MxK,row-major) @ Bt^T  (Bt is N x K row-major)
// 256 threads = 8 waves, 64x64 tile, K-step 64. Staging writes fragment-
// swizzled LDS via b128 global loads + b128 LDS stores.
// ---------------------------------------------------------------------------
template <int EPI>
__launch_bounds__(256)
__global__ void gemm_bf16_kernel(const bf16* __restrict__ A,
                                 const bf16* __restrict__ Bt,
                                 float* __restrict__ C, int M, int N, int K,
                                 const float* __restrict__ bias) {
  __shared__ __align__(16) bf16 As[8 * 32 * 16];  // subtile (kh*4+mt)
  __shared__ __align__(16) bf16 Bs[8 * 32 * 16];  // subtile (nt*2+kh)
  const int tid = threadIdx.x, lane = tid & 31, wv = tid >> 5;
  const int n0 = blockIdx.x * 64, m0 = blockIdx.y * 64;
  const int mt = wv & 3, ntp = wv >> 2;
  v8f acc0 = {}; v8f acc1 = {};

  // staging role: one 32-byte fragment row of A and of B per thread
  const int st = tid >> 5;
  const int mtA = st & 3, khA = st >> 2;     // A subtile id st == khA*4 + mtA
  const int khB = st & 1, ntB = st >> 1;     // B subtile id st == ntB*2 + khB
  const int lm = lane & 15, lhi = lane >> 4;
  const bf16* Arow = A + (size_t)(m0 + mtA * 16 + lm) * K;
  const bf16* Brow = Bt + (size_t)(n0 + ntB * 16 + lm) * K;
  bf16* Adst = As + (st * 32 + lane) * 16;
  bf16* Bdst = Bs + (st * 32 + lane) * 16;

  for (int k0 = 0; k0 < K; k0 += 64) {
    const int kaA = k0 + khA * 32 + 8 * lhi;
    const int kaB = k0 + khB * 32 + 8 * lhi;
    if (k0 + 64 <= K) {
      *(uint4*)(Adst)     = *(const uint4*)(Arow + kaA);
      *(uint4*)(Adst + 8) = *(const uint4*)(Arow + kaA + 16);
      *(uint4*)(Bdst)     = *(const uint4*)(Brow + kaB);
      *(uint4*)(Bdst + 8) = *(const uint4*)(Brow + kaB + 16);
    } else {  // K tail (GATEK=1040): guarded scalar staging
#pragma unroll
      for (int e = 0; e < 8; ++e) {
        int k1 = kaA + e, k2 = kaA + 16 + e;
        Adst[e]     = (k1 < K) ? Arow[k1] : (bf16)0.f;
        Adst[8 + e] = (k2 < K) ? Arow[k2] : (bf16)0.f;
        int k3 = kaB + e, k4 = kaB + 16 + e;
        Bdst[e]     = (k3 < K) ? Brow[k3] : (bf16)0.f;
        Bdst[8 + e] = (k4 < K) ? Brow[k4] : (bf16)0.f;
      }
    }
    __syncthreads();
#pragma unroll
    for (int kh = 0; kh < 2; ++kh) {
      v16bf af = ld_frag(As + ((kh * 4 + mt) * 32 + lane) * 16);
      v16bf b0 = ld_frag(Bs + (((2 * ntp) * 2 + kh) * 32 + lane) * 16);
      v16bf b1 = ld_frag(Bs + (((2 * ntp + 1) * 2 + kh) * 32 + lane) * 16);
      acc0 = wmma_bf16(af, b0, acc0);
      acc1 = wmma_bf16(af, b1, acc1);
    }
    __syncthreads();
  }
  const int n = lane & 15, hi = lane >> 4;
#pragma unroll
  for (int i = 0; i < 8; ++i) {
    int gm = m0 + mt * 16 + i + 8 * hi;
    int gc0 = n0 + (2 * ntp) * 16 + n, gc1 = gc0 + 16;
    float x0 = acc0[i], x1 = acc1[i];
    if (EPI == 1) {  // bias + exact gelu
      x0 += bias[gc0]; x1 += bias[gc1];
      x0 = 0.5f * x0 * (1.f + erff(x0 * 0.70710678f));
      x1 = 0.5f * x1 * (1.f + erff(x1 * 0.70710678f));
    }
    C[(size_t)gm * N + gc0] = x0;
    C[(size_t)gm * N + gc1] = x1;
  }
}

// ---------------------------------------------------------------------------
// Depthwise causal FIR conv over (BL, L, D=1024) channels, optional SiLU.
// ---------------------------------------------------------------------------
__global__ void dwconv_kernel(const float* __restrict__ x,
                              const float* __restrict__ w,
                              float* __restrict__ y, int BL, int L, int Kt,
                              int silu) {
  size_t idx = (size_t)blockIdx.x * blockDim.x + threadIdx.x;
  size_t total = (size_t)BL * HIDSZ;
  if (idx >= total) return;
  int d = (int)(idx & (HIDSZ - 1));
  size_t row = idx >> 10;
  int t = (int)(row % L);
  float s = 0.f;
  for (int j = 0; j < Kt; ++j) {
    int tt = t - (Kt - 1) + j;
    if (tt >= 0) s += x[(row - (Kt - 1) + j) * HIDSZ + d] * w[d * Kt + j];
  }
  if (silu) s = s / (1.f + expf(-s));
  y[idx] = s;
}

// ---------------------------------------------------------------------------
// beta = sigmoid(hs @ Wb)   (BL x 4)
// ---------------------------------------------------------------------------
__global__ void beta_kernel(const float* __restrict__ hs,
                            const float* __restrict__ Wb,
                            float* __restrict__ beta, int BL) {
  int idx = blockIdx.x * blockDim.x + threadIdx.x;
  if (idx >= BL * NHEAD) return;
  int h = idx & 3, row = idx >> 2;
  float s = 0.f;
  for (int i = 0; i < HIDSZ; ++i)
    s += hs[(size_t)row * HIDSZ + i] * Wb[i * NHEAD + h];
  beta[idx] = 1.f / (1.f + expf(-s));
}

// ---------------------------------------------------------------------------
// Chunked delta rule. One workgroup (8 waves) per (b,h). bf16 working state
// fragment-swizzled in LDS (~312KB of the 320KB WGP LDS, dynamic); fp32 master
// state in global (L2-resident). All S-sized matmuls are WMMA.
// ---------------------------------------------------------------------------
#define DELTA_LDS_BYTES 319488

__launch_bounds__(256)
__global__ void delta_kernel(const float* __restrict__ qg,
                             const float* __restrict__ kg,
                             const float* __restrict__ vg,
                             const float* __restrict__ betag,
                             float* __restrict__ Sg, float* __restrict__ dout,
                             int L) {
  extern __shared__ char smem[];
  bf16* S_sw   = (bf16*)smem;            // 256x256, B-swizzled (K=dk, N=dv)
  bf16* qn_bf  = S_sw + 65536;           // 32x256 row-major (VALU)
  bf16* kn_bf  = qn_bf + 8192;           // 32x256 row-major (VALU)
  bf16* kb_bf  = kn_bf + 8192;           // 32x256 row-major (VALU)
  bf16* qn_sw  = kb_bf + 8192;           // A-swizzled (M=32, K=256)
  bf16* kn_sw  = qn_sw + 8192;           // A-swizzled kn^T (M=256, K=32)
  bf16* w_sw   = kn_sw + 8192;           // A-swizzled (M=32, K=256)
  bf16* u_sw   = w_sw + 8192;            // B-swizzled (K=32, N=256)
  float* u_f   = (float*)(u_sw + 8192);  // 32x256
  float* o_f   = u_f + 8192;             // 32x256 (also stages v*beta)
  float* ainv  = o_f + 8192;             // 32x32
  float* loc   = ainv + 1024;            // 32x32

  const int tid = threadIdx.x, lane = tid & 31, wv = tid >> 5;
  const int bh = blockIdx.x, b = bh >> 2, h = bh & 3;
  float* Sglob = Sg + (size_t)bh * 65536;

  for (int i = tid; i < 65536; i += 256) { Sglob[i] = 0.f; S_sw[i] = (bf16)0.f; }
  __syncthreads();

  const int nchunks = L / 32;
  for (int ck = 0; ck < nchunks; ++ck) {
    const int t0 = ck * 32;
    // -- load q,k,v rows; L2-normalize q,k; scale by beta; swizzled copies --
#pragma unroll
    for (int rr = 0; rr < 4; ++rr) {
      int r = wv * 4 + rr;
      size_t rowoff = ((size_t)(b * L + t0 + r)) * HIDSZ + h * DHEAD;
      float qv[8], kv[8], qs = 0.f, ks = 0.f;
#pragma unroll
      for (int e = 0; e < 8; ++e) {
        qv[e] = qg[rowoff + lane + 32 * e]; qs += qv[e] * qv[e];
        kv[e] = kg[rowoff + lane + 32 * e]; ks += kv[e] * kv[e];
      }
#pragma unroll
      for (int m = 16; m; m >>= 1) { qs += __shfl_xor(qs, m); ks += __shfl_xor(ks, m); }
      float qsc = rsqrtf(qs + 1e-6f), ksc = rsqrtf(ks + 1e-6f);
      float bet = betag[(size_t)(b * L + t0 + r) * NHEAD + h];
#pragma unroll
      for (int e = 0; e < 8; ++e) {
        int d = lane + 32 * e;
        float qn = qv[e] * qsc, kn = kv[e] * ksc;
        qn_bf[r * 256 + d] = (bf16)qn;
        kn_bf[r * 256 + d] = (bf16)kn;
        kb_bf[r * 256 + d] = (bf16)(kn * bet);
        qn_sw[swz_addrA(r, d, 8)] = (bf16)qn;
        kn_sw[swz_addrA(d, r, 1)] = (bf16)kn;  // kn^T fragment layout
        o_f[r * 256 + d] = vg[rowoff + d] * bet;  // v*beta staged
      }
    }
    __syncthreads();
    // -- attn = -(kb @ kn^T), strict lower --
    for (int e = tid; e < 1024; e += 256) {
      int i = e >> 5, j = e & 31; float a = 0.f;
      if (j < i)
        for (int d = 0; d < 256; ++d)
          a -= (float)kb_bf[i * 256 + d] * (float)kn_bf[j * 256 + d];
      ainv[e] = a;
    }
    __syncthreads();
    // -- iterative row substitution -> inverse factor --
    for (int i = 1; i < 32; ++i) {
      float upd = 0.f;
      if (tid < i)
        for (int t = 0; t < i; ++t) upd += ainv[i * 32 + t] * ainv[t * 32 + tid];
      __syncthreads();
      if (tid < i) ainv[i * 32 + tid] += upd;
      __syncthreads();
    }
    if (tid < 32) ainv[tid * 32 + tid] += 1.f;
    __syncthreads();
    // -- u = inv @ (v*beta), w = inv @ kb --
    for (int e = tid; e < 8192; e += 256) {
      int i = e >> 8, d = e & 255;
      float su = 0.f, sw = 0.f;
#pragma unroll 8
      for (int j = 0; j < 32; ++j) {
        float a = ainv[i * 32 + j];
        su += a * o_f[j * 256 + d];
        sw += a * (float)kb_bf[j * 256 + d];
      }
      u_f[e] = su;
      w_sw[swz_addrA(i, d, 8)] = (bf16)sw;
    }
    __syncthreads();
    // -- attn_loc = (qn @ kn^T), inclusive lower --
    for (int e = tid; e < 1024; e += 256) {
      int i = e >> 5, j = e & 31; float a = 0.f;
      if (j <= i)
        for (int d = 0; d < 256; ++d)
          a += (float)qn_bf[i * 256 + d] * (float)kn_bf[j * 256 + d];
      loc[e] = a;
    }
    // -- u -= w @ S   (WMMA, 2x16 tiles, 4/wave) --
    for (int ti = 0; ti < 4; ++ti) {
      int tix = wv * 4 + ti, mt = tix >> 4, nt = tix & 15;
      v8f acc = {};
#pragma unroll
      for (int kt = 0; kt < 8; ++kt) {
        v16bf af  = ld_frag(w_sw + ((mt * 8 + kt) * 32 + lane) * 16);
        v16bf bfr = ld_frag(S_sw + ((kt * 16 + nt) * 32 + lane) * 16);
        acc = wmma_bf16(af, bfr, acc);
      }
      int n = lane & 15, hi = lane >> 4;
#pragma unroll
      for (int i2 = 0; i2 < 8; ++i2)
        u_f[(mt * 16 + i2 + 8 * hi) * 256 + nt * 16 + n] -= acc[i2];
    }
    __syncthreads();
    for (int e = tid; e < 8192; e += 256) {
      int i = e >> 8, d = e & 255;
      u_sw[swz_addrB(i, d, 16)] = (bf16)u_f[e];
    }
    __syncthreads();
    // -- o = qn @ S   (WMMA) --
    for (int ti = 0; ti < 4; ++ti) {
      int tix = wv * 4 + ti, mt = tix >> 4, nt = tix & 15;
      v8f acc = {};
#pragma unroll
      for (int kt = 0; kt < 8; ++kt) {
        v16bf af  = ld_frag(qn_sw + ((mt * 8 + kt) * 32 + lane) * 16);
        v16bf bfr = ld_frag(S_sw + ((kt * 16 + nt) * 32 + lane) * 16);
        acc = wmma_bf16(af, bfr, acc);
      }
      int n = lane & 15, hi = lane >> 4;
#pragma unroll
      for (int i2 = 0; i2 < 8; ++i2)
        o_f[(mt * 16 + i2 + 8 * hi) * 256 + nt * 16 + n] = acc[i2];
    }
    __syncthreads();
    // -- o += attn_loc @ u ; store chunk output --
    for (int e = tid; e < 8192; e += 256) {
      int i = e >> 8, d = e & 255;
      float s = 0.f;
#pragma unroll 8
      for (int j = 0; j < 32; ++j) s += loc[i * 32 + j] * u_f[j * 256 + d];
      dout[(((size_t)(b * L + t0 + i)) * NHEAD + h) * DHEAD + d] = o_f[e] + s;
    }
    // -- S += kn^T @ u   (WMMA, 16x16 tiles, 32/wave, single K=32 step) --
    for (int ti = 0; ti < 32; ++ti) {
      int tix = wv * 32 + ti, mt = tix >> 4, nt = tix & 15;
      int n = lane & 15, hi = lane >> 4;
      v8f acc;
#pragma unroll
      for (int i2 = 0; i2 < 8; ++i2)
        acc[i2] = Sglob[(mt * 16 + i2 + 8 * hi) * 256 + nt * 16 + n];
      v16bf af  = ld_frag(kn_sw + (mt * 32 + lane) * 16);
      v16bf bfr = ld_frag(u_sw + (nt * 32 + lane) * 16);
      acc = wmma_bf16(af, bfr, acc);
#pragma unroll
      for (int i2 = 0; i2 < 8; ++i2) {
        float vv = acc[i2];
        int dk = mt * 16 + i2 + 8 * hi, dv = nt * 16 + n;
        Sglob[dk * 256 + dv] = vv;
        S_sw[swz_addrB(dk, dv, 16)] = (bf16)vv;
      }
    }
    __syncthreads();
  }
}

// ---------------------------------------------------------------------------
// Per-(token,head) branch statistics + gate-input assembly (bf16 rows of 1040)
// ---------------------------------------------------------------------------
__global__ void stats_gate_kernel(const bf16* __restrict__ hs_bf,
                                  const float* __restrict__ ls,
                                  const float* __restrict__ ll,
                                  const float* __restrict__ dl,
                                  const float* __restrict__ vv,
                                  bf16* __restrict__ gate, int BL) {
  int lane = threadIdx.x & 31, wv = threadIdx.x >> 5;
  int g = blockIdx.x * 8 + wv;
  if (g >= BL * NHEAD) return;
  int row = g >> 2, h = g & 3;
  bf16* grow = gate + (size_t)g * GATEK;
  const bf16* hrow = hs_bf + (size_t)row * HIDSZ;
  for (int i = lane; i < HIDSZ; i += 32) grow[i] = hrow[i];
  const float* sig[4] = {ls + (size_t)row * HIDSZ + h * DHEAD,
                         ll + (size_t)row * HIDSZ + h * DHEAD,
                         dl + ((size_t)row * NHEAD + h) * DHEAD,
                         vv + (size_t)row * HIDSZ + h * DHEAD};
#pragma unroll
  for (int s = 0; s < 4; ++s) {
    float s1 = 0.f, s2 = 0.f, sa = 0.f;
#pragma unroll
    for (int e = 0; e < 8; ++e) {
      float x = sig[s][lane + 32 * e];
      s1 += x; s2 += x * x; sa += fabsf(x);
    }
#pragma unroll
    for (int m = 16; m; m >>= 1) {
      s1 += __shfl_xor(s1, m); s2 += __shfl_xor(s2, m); sa += __shfl_xor(sa, m);
    }
    if (lane == 0) {
      float mean = s1 * (1.f / 256.f);
      grow[HIDSZ + s * 4 + 0] = (bf16)mean;
      grow[HIDSZ + s * 4 + 1] = (bf16)(s2 * (1.f / 256.f) - mean * mean);
      grow[HIDSZ + s * 4 + 2] = (bf16)(sa * (1.f / 256.f));
      grow[HIDSZ + s * 4 + 3] = (bf16)sqrtf(s2);
    }
  }
}

// ---------------------------------------------------------------------------
// logits = h1 @ mlp_w2 + b2   (rows x 4)
// ---------------------------------------------------------------------------
__global__ void logits_kernel(const float* __restrict__ h1,
                              const float* __restrict__ W2,
                              const float* __restrict__ b2,
                              float* __restrict__ lg, int rows) {
  int idx = blockIdx.x * blockDim.x + threadIdx.x;
  if (idx >= rows * 4) return;
  int c = idx & 3, g = idx >> 2;
  float s = b2[c];
  for (int i = 0; i < HIDSZ; ++i) s += h1[(size_t)g * HIDSZ + i] * W2[i * 4 + c];
  lg[idx] = s;
}

// ---------------------------------------------------------------------------
// Softmax(temp)+floor fusion of 4 branches, per-head RMSNorm, emit bf16 rows.
// ---------------------------------------------------------------------------
__global__ void combine_kernel(const float* __restrict__ lg,
                               const float* __restrict__ log_temp,
                               const float* __restrict__ onw,
                               const float* __restrict__ ls,
                               const float* __restrict__ ll,
                               const float* __restrict__ dl,
                               const float* __restrict__ vv,
                               bf16* __restrict__ obf, int BL) {
  int lane = threadIdx.x & 31, wv = threadIdx.x >> 5;
  int g = blockIdx.x * 8 + wv;
  if (g >= BL * NHEAD) return;
  int row = g >> 2, h = g & 3;
  float invT = expf(-log_temp[h]);
  float l0 = lg[g * 4 + 0] * invT, l1 = lg[g * 4 + 1] * invT;
  float l2 = lg[g * 4 + 2] * invT, l3 = lg[g * 4 + 3] * invT;
  float mx = fmaxf(fmaxf(l0, l1), fmaxf(l2, l3));
  float e0 = expf(l0 - mx), e1 = expf(l1 - mx), e2 = expf(l2 - mx), e3 = expf(l3 - mx);
  float inv = 1.f / (e0 + e1 + e2 + e3);
  float p0 = FLOORW + 0.8f * e0 * inv, p1 = FLOORW + 0.8f * e1 * inv;
  float p2 = FLOORW + 0.8f * e2 * inv, p3 = FLOORW + 0.8f * e3 * inv;
  const float* a0 = ls + (size_t)row * HIDSZ + h * DHEAD;
  const float* a1 = ll + (size_t)row * HIDSZ + h * DHEAD;
  const float* a2 = dl + ((size_t)row * NHEAD + h) * DHEAD;
  const float* a3 = vv + (size_t)row * HIDSZ + h * DHEAD;
  float vals[8], ssq = 0.f;
#pragma unroll
  for (int e = 0; e < 8; ++e) {
    int d = lane + 32 * e;
    float o = p0 * a0[d] + p1 * a1[d] + p2 * a2[d] + p3 * a3[d];
    vals[e] = o; ssq += o * o;
  }
#pragma unroll
  for (int m = 16; m; m >>= 1) ssq += __shfl_xor(ssq, m);
  float r = rsqrtf(ssq * (1.f / 256.f) + 1e-5f);
#pragma unroll
  for (int e = 0; e < 8; ++e) {
    int d = lane + 32 * e;
    obf[(size_t)row * HIDSZ + h * DHEAD + d] = (bf16)(vals[e] * r * onw[d]);
  }
}

// ---------------------------------------------------------------------------
// Host launcher
// ---------------------------------------------------------------------------
extern "C" void kernel_launch(void* const* d_in, const int* in_sizes, int n_in,
                              void* d_out, int out_size, void* d_ws, size_t ws_size,
                              hipStream_t stream) {
  (void)n_in; (void)out_size; (void)ws_size;
  const float* hs   = (const float*)d_in[0];
  const float* Wq   = (const float*)d_in[1];
  const float* Wk   = (const float*)d_in[2];
  const float* Wv   = (const float*)d_in[3];
  const float* Wb   = (const float*)d_in[4];
  const float* cqw  = (const float*)d_in[5];
  const float* ckw  = (const float*)d_in[6];
  const float* cvw  = (const float*)d_in[7];
  const float* fsw  = (const float*)d_in[8];
  const float* flw  = (const float*)d_in[9];
  const float* w1   = (const float*)d_in[10];
  const float* b1   = (const float*)d_in[11];
  const float* w2   = (const float*)d_in[12];
  const float* b2   = (const float*)d_in[13];
  const float* ltmp = (const float*)d_in[14];
  const float* onw  = (const float*)d_in[15];
  const float* Wo   = (const float*)d_in[16];
  float* out = (float*)d_out;

  const int BL = in_sizes[0] / HIDSZ;   // B*L = 8192
  const int L  = BL / 2;                // B = 2
  const int ROWS = BL * NHEAD;          // 32768

  char* wsp = (char*)d_ws;
  size_t off = 0;
  auto alloc = [&](size_t bytes) -> void* {
    void* p = wsp + off;
    off += (bytes + 255) & ~(size_t)255;
    return p;
  };
  bf16*  hs_bf   = (bf16*)alloc((size_t)BL * HIDSZ * 2);
  bf16*  wbuf    = (bf16*)alloc((size_t)GATEK * HIDSZ * 2);  // reused (transposed) weights
  float* linbuf  = (float*)alloc((size_t)BL * HIDSZ * 4);    // reused GEMM out
  float* qbuf    = (float*)alloc((size_t)BL * HIDSZ * 4);
  float* kbuf    = (float*)alloc((size_t)BL * HIDSZ * 4);
  float* vbuf    = (float*)alloc((size_t)BL * HIDSZ * 4);
  float* betab   = (float*)alloc((size_t)BL * NHEAD * 4);
  float* Sstate  = (float*)alloc((size_t)8 * 65536 * 4);
  float* deltab  = (float*)alloc((size_t)BL * HIDSZ * 4);
  float* lsbuf   = (float*)alloc((size_t)BL * HIDSZ * 4);
  float* llbuf   = (float*)alloc((size_t)BL * HIDSZ * 4);
  bf16*  gatebf  = (bf16*)alloc((size_t)ROWS * GATEK * 2);
  float* h1buf   = (float*)alloc((size_t)ROWS * HIDSZ * 4);
  float* lgbuf   = (float*)alloc((size_t)ROWS * 4 * 4);
  bf16*  obf     = (bf16*)alloc((size_t)BL * HIDSZ * 2);

  const int ewB = 2048;
  dim3 blk(256);

  // hs -> bf16
  f2bf_kernel<<<ewB, blk, 0, stream>>>(hs, hs_bf, (size_t)BL * HIDSZ);

  // q/k/v projections (WMMA, transposed weights) + causal conv4 + SiLU
  dim3 gProj(HIDSZ / 64, BL / 64);
  int convB = (int)(((size_t)BL * HIDSZ + 255) / 256);
  int wtB = (int)(((size_t)HIDSZ * HIDSZ + 255) / 256);
  const float* projW[3] = {Wq, Wk, Wv};
  const float* convW[3] = {cqw, ckw, cvw};
  float* dst[3] = {qbuf, kbuf, vbuf};
  for (int s = 0; s < 3; ++s) {
    f2bf_t_kernel<<<wtB, blk, 0, stream>>>(projW[s], wbuf, HIDSZ, HIDSZ);
    gemm_bf16_kernel<0><<<gProj, blk, 0, stream>>>(hs_bf, wbuf, linbuf, BL, HIDSZ,
                                                   HIDSZ, nullptr);
    dwconv_kernel<<<convB, blk, 0, stream>>>(linbuf, convW[s], dst[s], BL, L, 4, 1);
  }

  // beta
  beta_kernel<<<(BL * NHEAD + 255) / 256, blk, 0, stream>>>(hs, Wb, betab, BL);

  // chunked delta rule (WMMA state matmuls)
  delta_kernel<<<8, blk, DELTA_LDS_BYTES, stream>>>(qbuf, kbuf, vbuf, betab,
                                                    Sstate, deltab, L);

  // FIR branches on v_direct
  dwconv_kernel<<<convB, blk, 0, stream>>>(vbuf, fsw, lsbuf, BL, L, 5, 0);
  dwconv_kernel<<<convB, blk, 0, stream>>>(vbuf, flw, llbuf, BL, L, 64, 0);

  // stats + gate input rows
  stats_gate_kernel<<<ROWS / 8, blk, 0, stream>>>(hs_bf, lsbuf, llbuf, deltab,
                                                  vbuf, gatebf, BL);

  // gate MLP layer 1 (WMMA + fused bias/gelu), transposed weights
  int w1B = (int)(((size_t)GATEK * HIDSZ + 255) / 256);
  f2bf_t_kernel<<<w1B, blk, 0, stream>>>(w1, wbuf, GATEK, HIDSZ);
  dim3 gMlp(HIDSZ / 64, ROWS / 64);
  gemm_bf16_kernel<1><<<gMlp, blk, 0, stream>>>(gatebf, wbuf, h1buf, ROWS, HIDSZ,
                                                GATEK, b1);

  // logits, fusion, RMSNorm
  logits_kernel<<<(ROWS * 4 + 255) / 256, blk, 0, stream>>>(h1buf, w2, b2, lgbuf,
                                                            ROWS);
  combine_kernel<<<ROWS / 8, blk, 0, stream>>>(lgbuf, ltmp, onw, lsbuf, llbuf,
                                               deltab, vbuf, obf, BL);

  // output projection (WMMA) -> d_out (fp32)
  f2bf_t_kernel<<<wtB, blk, 0, stream>>>(Wo, wbuf, HIDSZ, HIDSZ);
  gemm_bf16_kernel<0><<<gProj, blk, 0, stream>>>(obf, wbuf, out, BL, HIDSZ, HIDSZ,
                                                 nullptr);
}
